// TransformerEncoderLayer_8881992368674
// MI455X (gfx1250) — compile-verified, matrix operand order checked by reading, not run on hardware
//
#include <hip/hip_runtime.h>
#include <hip/hip_bf16.h>
#include <stdint.h>

// ---------------------------------------------------------------------------
// Transformer encoder layer on gfx1250 (MI455X), bf16 WMMA everywhere.
// B=2 S=2048 D=1024 H=4096, 16 heads x 64.
// GEMMs: v_wmma_f32_16x16x32_bf16, staging via global_load_async_to_lds_b128
// with double-buffered LDS (one barrier per k-step, ASYNCcnt-tracked copies).
// ---------------------------------------------------------------------------

typedef __bf16 bf16_t;
typedef __attribute__((ext_vector_type(16))) __bf16         v16bf;
typedef __attribute__((ext_vector_type(8)))  float          v8f;
typedef __attribute__((ext_vector_type(8)))  unsigned short ushort8; // 16 bytes

__device__ __forceinline__ v8f zero8() {
    v8f z;
#pragma unroll
    for (int i = 0; i < 8; ++i) z[i] = 0.0f;
    return z;
}

// Async copy 16B / 32B from global memory directly into LDS (ASYNCcnt).
// IOFFSET is added to BOTH the LDS and the global address (ISA 10.x async),
// so one base pair + offset:16 covers a 32B chunk.
__device__ __forceinline__ void async_b128(uint32_t lds_addr, const void* gaddr) {
    asm volatile("global_load_async_to_lds_b128 %0, %1, off"
                 :: "v"(lds_addr), "v"((uint64_t)(uintptr_t)gaddr)
                 : "memory");
}
__device__ __forceinline__ void async_b128x2(uint32_t lds_addr, const void* gaddr) {
    asm volatile("global_load_async_to_lds_b128 %0, %1, off"
                 :: "v"(lds_addr), "v"((uint64_t)(uintptr_t)gaddr)
                 : "memory");
    asm volatile("global_load_async_to_lds_b128 %0, %1, off offset:16"
                 :: "v"(lds_addr), "v"((uint64_t)(uintptr_t)gaddr)
                 : "memory");
}
__device__ __forceinline__ void wait_async0() {
    asm volatile("s_wait_asynccnt 0x0" ::: "memory");
}

__device__ __forceinline__ uint32_t lds_addr_of(const void* p) {
    return (uint32_t)(uintptr_t)p;  // LDS aperture occupies high 32 bits only
}

// Load a 16-element bf16 fragment as two 16B chunks (A or B operand layout:
// lane holds row (lane&15); k indices kb..kb+7 and kb+16..kb+23, kb=(lane>>4)*8)
__device__ __forceinline__ v16bf frag_ld(const bf16_t* p0, const bf16_t* p1) {
    union {
        ushort8 u[2];
        v16bf   v;
    } t;
    t.u[0] = *(const ushort8*)p0;
    t.u[1] = *(const ushort8*)p1;
    return t.v;
}

__device__ __forceinline__ v8f wmma_bf16(v16bf a, v16bf b, v8f c) {
    return __builtin_amdgcn_wmma_f32_16x16x32_bf16(
        /*neg_a=*/false, a, /*neg_b=*/false, b,
        /*c_mod=*/(short)0, c, /*reuse_a=*/false, /*reuse_b=*/false);
}

// ---------------------------------------------------------------------------
// fp32 -> bf16 conversion (weights)
// ---------------------------------------------------------------------------
__global__ __launch_bounds__(256) void f32_to_bf16_kernel(const float* __restrict__ in,
                                                          bf16_t* __restrict__ out,
                                                          long n) {
    long i = ((long)blockIdx.x * 256 + threadIdx.x) * 4;
    if (i + 3 < n) {
        float4 f = *(const float4*)(in + i);
        out[i + 0] = (bf16_t)f.x;
        out[i + 1] = (bf16_t)f.y;
        out[i + 2] = (bf16_t)f.z;
        out[i + 3] = (bf16_t)f.w;
    }
}

// ---------------------------------------------------------------------------
// LayerNorm over D=1024; one block (8 waves) per row. Emits fp32 + bf16.
// ---------------------------------------------------------------------------
__global__ __launch_bounds__(256) void layernorm_kernel(const float* __restrict__ x,
                                                        const float* __restrict__ g,
                                                        const float* __restrict__ b,
                                                        float* __restrict__ of,
                                                        bf16_t* __restrict__ ob) {
    const long row = blockIdx.x;
    const float* xr = x + row * 1024;
    const int tid = threadIdx.x;

    float v[4], s = 0.f, s2 = 0.f;
#pragma unroll
    for (int i = 0; i < 4; ++i) {
        v[i] = xr[tid + 256 * i];
        s += v[i];
        s2 += v[i] * v[i];
    }
#pragma unroll
    for (int off = 16; off >= 1; off >>= 1) {
        s += __shfl_xor(s, off, 32);
        s2 += __shfl_xor(s2, off, 32);
    }
    __shared__ float ws1[8], ws2[8];
    if ((tid & 31) == 0) {
        ws1[tid >> 5] = s;
        ws2[tid >> 5] = s2;
    }
    __syncthreads();
    s = 0.f;
    s2 = 0.f;
#pragma unroll
    for (int i = 0; i < 8; ++i) {
        s += ws1[i];
        s2 += ws2[i];
    }
    const float mu   = s * (1.0f / 1024.0f);
    const float var  = s2 * (1.0f / 1024.0f) - mu * mu;
    const float rstd = rsqrtf(var + 1e-5f);
#pragma unroll
    for (int i = 0; i < 4; ++i) {
        int c   = tid + 256 * i;
        float y = (v[i] - mu) * rstd * g[c] + b[c];
        of[row * 1024 + c] = y;
        if (ob) ob[row * 1024 + c] = (bf16_t)y;
    }
}

// ---------------------------------------------------------------------------
// Generic bf16 WMMA GEMM:  C[N,M] = A[N,K] * B[M,K]^T (+bias[M]) (+resid[N,M])
// optional exact GELU; fp32 and/or bf16 outputs.
// BM=BN=128, BK=32, 256 threads (8 waves), wave tile 32x64.
// Double-buffered LDS; async copies overlap next tile with current WMMAs.
// ---------------------------------------------------------------------------
#define GEMM_BM 128
#define GEMM_BK 32
#define GEMM_LS 40  // padded LDS row stride (bf16 elems)

__global__ __launch_bounds__(256) void gemm_bf16_kernel(const bf16_t* __restrict__ A,
                                                        const bf16_t* __restrict__ B,
                                                        int K, int M,
                                                        const float* __restrict__ bias,
                                                        const float* __restrict__ resid,
                                                        float* __restrict__ Cf,
                                                        bf16_t* __restrict__ Cb,
                                                        int do_gelu) {
    __shared__ bf16_t Asm[2][GEMM_BM * GEMM_LS];
    __shared__ bf16_t Bsm[2][GEMM_BM * GEMM_LS];

    const int tid  = threadIdx.x;
    const int lane = tid & 31;
    const int wid  = tid >> 5;
    const int wm   = wid & 3;   // 4 waves along rows (32 rows each)
    const int wn   = wid >> 2;  // 2 waves along cols (64 cols each)
    const long n0  = (long)blockIdx.x * GEMM_BM;
    const long m0  = (long)blockIdx.y * GEMM_BM;

    v8f acc[2][4];
#pragma unroll
    for (int i = 0; i < 2; ++i)
#pragma unroll
        for (int j = 0; j < 4; ++j) acc[i][j] = zero8();

    // staging: thread t owns 32B of one row (row = t/2, 16 elems at col (t&1)*16)
    const int ldrow = tid >> 1;
    const int ldcol = (tid & 1) * 16;
    const bf16_t* Ag = A + (n0 + ldrow) * (long)K + ldcol;
    const bf16_t* Bg = B + (m0 + ldrow) * (long)K + ldcol;
    uint32_t aL[2], bL[2];
    aL[0] = lds_addr_of(&Asm[0][ldrow * GEMM_LS + ldcol]);
    aL[1] = lds_addr_of(&Asm[1][ldrow * GEMM_LS + ldcol]);
    bL[0] = lds_addr_of(&Bsm[0][ldrow * GEMM_LS + ldcol]);
    bL[1] = lds_addr_of(&Bsm[1][ldrow * GEMM_LS + ldcol]);

    const int frow = lane & 15;
    const int kb   = (lane >> 4) * 8;

    // prologue: stage first k-tile into buffer 0
    async_b128x2(aL[0], Ag);
    async_b128x2(bL[0], Bg);

    int buf = 0;
    for (int k0 = 0; k0 < K; k0 += GEMM_BK) {
        wait_async0();
        __syncthreads();
        if (k0 + GEMM_BK < K) {
            async_b128x2(aL[buf ^ 1], Ag + k0 + GEMM_BK);
            async_b128x2(bL[buf ^ 1], Bg + k0 + GEMM_BK);
        }

        v16bf af[2], bf[4];
#pragma unroll
        for (int im = 0; im < 2; ++im) {
            const bf16_t* ar = &Asm[buf][(32 * wm + 16 * im + frow) * GEMM_LS];
            af[im] = frag_ld(ar + kb, ar + kb + 16);
        }
#pragma unroll
        for (int jn = 0; jn < 4; ++jn) {
            const bf16_t* br = &Bsm[buf][(64 * wn + 16 * jn + frow) * GEMM_LS];
            bf[jn] = frag_ld(br + kb, br + kb + 16);
        }
#pragma unroll
        for (int im = 0; im < 2; ++im)
#pragma unroll
            for (int jn = 0; jn < 4; ++jn)
                acc[im][jn] = wmma_bf16(af[im], bf[jn], acc[im][jn]);
        buf ^= 1;
    }

    // epilogue: C layout -> row = i + 8*(lane>=16), col = lane&15 (per 16x16 tile)
    const int hw = lane >> 4;
#pragma unroll
    for (int im = 0; im < 2; ++im) {
#pragma unroll
        for (int jn = 0; jn < 4; ++jn) {
            const long c   = m0 + 64 * wn + 16 * jn + frow;
            const float bv = bias ? bias[c] : 0.0f;
#pragma unroll
            for (int i = 0; i < 8; ++i) {
                const long r = n0 + 32 * wm + 16 * im + i + 8 * hw;
                float val    = acc[im][jn][i] + bv;
                if (resid) val += resid[r * (long)M + c];
                if (do_gelu) val = 0.5f * val * (1.0f + erff(val * 0.70710678f));
                if (Cf) Cf[r * (long)M + c] = val;
                if (Cb) Cb[r * (long)M + c] = (bf16_t)val;
            }
        }
    }
}

// ---------------------------------------------------------------------------
// Flash attention: head_dim=64, scale=1/8. Grid: (S/128, B*16 heads).
// 8 waves; each wave owns 16 query rows, streams keys in 32-wide tiles.
// K tile staged with async-to-LDS; V staged transposed through registers.
// ---------------------------------------------------------------------------
#define ATT_TK 32
#define ATT_HD 64

__global__ __launch_bounds__(256) void attn_kernel(const bf16_t* __restrict__ Q,
                                                   const bf16_t* __restrict__ K,
                                                   const bf16_t* __restrict__ V,
                                                   bf16_t* __restrict__ O,
                                                   int S, int D) {
    __shared__ bf16_t Ksm[ATT_TK * 72];        // 32 keys x 64 hd (padded)
    __shared__ bf16_t Vts[ATT_HD * 40];        // transposed: 64 hd x 32 keys (padded)
    __shared__ bf16_t Psm[8 * 16 * 40];        // per-wave P relayout scratch

    const int tid  = threadIdx.x;
    const int lane = tid & 31;
    const int wid  = tid >> 5;
    const int bh   = blockIdx.y;
    const int b    = bh >> 4;
    const int h    = bh & 15;
    const int q0   = blockIdx.x * 128 + wid * 16;

    const bf16_t* qbase = Q + ((long)b * S) * D + h * ATT_HD;
    const bf16_t* kbase = K + ((long)b * S) * D + h * ATT_HD;
    const bf16_t* vbase = V + ((long)b * S) * D + h * ATT_HD;

    const int frow = lane & 15;
    const int kb   = (lane >> 4) * 8;
    const int hw   = lane >> 4;

    // Q fragments for this wave's 16 rows (hd split into two K=32 chunks)
    v16bf qf[2];
    {
        const bf16_t* qr = qbase + (long)(q0 + frow) * D;
        qf[0] = frag_ld(qr + kb, qr + kb + 16);
        qf[1] = frag_ld(qr + 32 + kb, qr + 32 + kb + 16);
    }

    v8f oacc[4];
#pragma unroll
    for (int j = 0; j < 4; ++j) oacc[j] = zero8();
    float m[8], l[8];
#pragma unroll
    for (int i = 0; i < 8; ++i) {
        m[i] = -1e30f;
        l[i] = 0.0f;
    }

    const int krow = tid >> 3;          // 0..31 keys
    const int kcol = (tid & 7) * 8;     // hd chunk
    const int vkey = tid & 31;
    const int vhd  = (tid >> 5) * 8;
    const uint32_t kLds = lds_addr_of(&Ksm[krow * 72 + kcol]);

    for (int kt = 0; kt < S; kt += ATT_TK) {
        // stage K tile (async copy engine) and V tile (transposed) into LDS
        async_b128(kLds, kbase + (long)(kt + krow) * D + kcol);
        {
            union {
                ushort8 u;
                bf16_t  e[8];
            } vv;
            vv.u = *(const ushort8*)(vbase + (long)(kt + vkey) * D + vhd);
#pragma unroll
            for (int i = 0; i < 8; ++i) Vts[(vhd + i) * 40 + vkey] = vv.e[i];
        }
        if (kt + ATT_TK < S)
            __builtin_prefetch((const void*)(vbase + (long)(kt + ATT_TK + vkey) * D + vhd), 0, 3);
        wait_async0();
        __syncthreads();

        // scores: 16 q rows x 32 keys
        v8f sc[2];
        sc[0] = zero8();
        sc[1] = zero8();
#pragma unroll
        for (int j = 0; j < 2; ++j) {
            const bf16_t* kr = &Ksm[(16 * j + frow) * 72];
            v16bf kf0 = frag_ld(kr + kb, kr + kb + 16);
            v16bf kf1 = frag_ld(kr + 32 + kb, kr + 32 + kb + 16);
            sc[j] = wmma_bf16(qf[0], kf0, sc[j]);
            sc[j] = wmma_bf16(qf[1], kf1, sc[j]);
        }
#pragma unroll
        for (int j = 0; j < 2; ++j)
#pragma unroll
            for (int i = 0; i < 8; ++i) sc[j][i] *= 0.125f;

        // online softmax (row reductions stay inside 16-lane halves)
        float rm[8];
#pragma unroll
        for (int i = 0; i < 8; ++i) rm[i] = fmaxf(sc[0][i], sc[1][i]);
#pragma unroll
        for (int off = 1; off < 16; off <<= 1)
#pragma unroll
            for (int i = 0; i < 8; ++i) rm[i] = fmaxf(rm[i], __shfl_xor(rm[i], off, 32));

        float alpha[8], rs[8];
#pragma unroll
        for (int i = 0; i < 8; ++i) {
            float mn = fmaxf(m[i], rm[i]);
            alpha[i] = __expf(m[i] - mn);
            m[i]     = mn;
            rs[i]    = 0.0f;
        }
#pragma unroll
        for (int j = 0; j < 2; ++j)
#pragma unroll
            for (int i = 0; i < 8; ++i) {
                float p  = __expf(sc[j][i] - m[i]);
                sc[j][i] = p;
                rs[i] += p;
            }
#pragma unroll
        for (int off = 1; off < 16; off <<= 1)
#pragma unroll
            for (int i = 0; i < 8; ++i) rs[i] += __shfl_xor(rs[i], off, 32);
#pragma unroll
        for (int i = 0; i < 8; ++i) l[i] = l[i] * alpha[i] + rs[i];
#pragma unroll
        for (int j = 0; j < 4; ++j)
#pragma unroll
            for (int i = 0; i < 8; ++i) oacc[j][i] *= alpha[i];

        // relayout P (C layout -> A fragment layout) through per-wave LDS
        bf16_t* pb = &Psm[wid * 16 * 40];
#pragma unroll
        for (int j = 0; j < 2; ++j)
#pragma unroll
            for (int i = 0; i < 8; ++i)
                pb[(i + 8 * hw) * 40 + 16 * j + frow] = (bf16_t)sc[j][i];

        {
            const bf16_t* pr = pb + frow * 40;
            v16bf pf = frag_ld(pr + kb, pr + kb + 16);
#pragma unroll
            for (int jj = 0; jj < 4; ++jj) {
                const bf16_t* vr = &Vts[(16 * jj + frow) * 40];
                v16bf vf = frag_ld(vr + kb, vr + kb + 16);
                oacc[jj] = wmma_bf16(pf, vf, oacc[jj]);
            }
        }
        __syncthreads();
    }

    // normalize and write O (bf16, (b,s,h,d) packed as N x D rows)
    float inv[8];
#pragma unroll
    for (int i = 0; i < 8; ++i) inv[i] = 1.0f / l[i];
    bf16_t* obase = O + ((long)b * S + q0) * D + h * ATT_HD;
#pragma unroll
    for (int jj = 0; jj < 4; ++jj)
#pragma unroll
        for (int i = 0; i < 8; ++i)
            obase[(long)(i + 8 * hw) * D + 16 * jj + frow] = (bf16_t)(oacc[jj][i] * inv[i]);
}

// ---------------------------------------------------------------------------
// Host orchestration
// ---------------------------------------------------------------------------
extern "C" void kernel_launch(void* const* d_in, const int* in_sizes, int n_in,
                              void* d_out, int out_size, void* d_ws, size_t ws_size,
                              hipStream_t stream) {
    const float* x   = (const float*)d_in[0];
    const float* wq  = (const float*)d_in[1];
    const float* bq  = (const float*)d_in[2];
    const float* wk  = (const float*)d_in[3];
    const float* bk  = (const float*)d_in[4];
    const float* wv  = (const float*)d_in[5];
    const float* bv  = (const float*)d_in[6];
    const float* wo  = (const float*)d_in[7];
    const float* bo  = (const float*)d_in[8];
    const float* g1  = (const float*)d_in[9];
    const float* be1 = (const float*)d_in[10];
    const float* w1  = (const float*)d_in[11];
    const float* b1  = (const float*)d_in[12];
    const float* w2  = (const float*)d_in[13];
    const float* b2  = (const float*)d_in[14];
    const float* g2  = (const float*)d_in[15];
    const float* be2 = (const float*)d_in[16];

    const long N = 4096;  // B*S
    const int  D = 1024, H = 4096, S = 2048;

    char* p = (char*)d_ws;
    auto alloc = [&](size_t bytes) -> void* {
        void* r = (void*)p;
        p += (bytes + 255) & ~(size_t)255;
        return r;
    };
    float*  xn_f = (float*)alloc(N * D * 4);
    bf16_t* xn_b = (bf16_t*)alloc(N * D * 2);
    bf16_t* wq_b = (bf16_t*)alloc((long)D * D * 2);
    bf16_t* wk_b = (bf16_t*)alloc((long)D * D * 2);
    bf16_t* wv_b = (bf16_t*)alloc((long)D * D * 2);
    bf16_t* wo_b = (bf16_t*)alloc((long)D * D * 2);
    bf16_t* w1_b = (bf16_t*)alloc((long)H * D * 2);
    bf16_t* w2_b = (bf16_t*)alloc((long)D * H * 2);
    bf16_t* q_b  = (bf16_t*)alloc(N * D * 2);
    bf16_t* k_b  = (bf16_t*)alloc(N * D * 2);
    bf16_t* v_b  = (bf16_t*)alloc(N * D * 2);
    bf16_t* at_b = (bf16_t*)alloc(N * D * 2);
    float*  y1_f = (float*)alloc(N * D * 4);
    float*  y2_f = (float*)alloc(N * D * 4);
    bf16_t* y2_b = (bf16_t*)alloc(N * D * 2);
    bf16_t* h_b  = (bf16_t*)alloc(N * (long)H * 2);

    // weight conversions
    f32_to_bf16_kernel<<<1024, 256, 0, stream>>>(wq, wq_b, (long)D * D);
    f32_to_bf16_kernel<<<1024, 256, 0, stream>>>(wk, wk_b, (long)D * D);
    f32_to_bf16_kernel<<<1024, 256, 0, stream>>>(wv, wv_b, (long)D * D);
    f32_to_bf16_kernel<<<1024, 256, 0, stream>>>(wo, wo_b, (long)D * D);
    f32_to_bf16_kernel<<<4096, 256, 0, stream>>>(w1, w1_b, (long)H * D);
    f32_to_bf16_kernel<<<4096, 256, 0, stream>>>(w2, w2_b, (long)D * H);

    // LN1: x -> xn (fp32 + bf16)
    layernorm_kernel<<<N, 256, 0, stream>>>(x, g1, be1, xn_f, xn_b);

    // QKV projections (bf16 out)
    dim3 blk(256);
    dim3 gDD(N / 128, D / 128);
    gemm_bf16_kernel<<<gDD, blk, 0, stream>>>(xn_b, wq_b, D, D, bq, nullptr, nullptr, q_b, 0);
    gemm_bf16_kernel<<<gDD, blk, 0, stream>>>(xn_b, wk_b, D, D, bk, nullptr, nullptr, k_b, 0);
    gemm_bf16_kernel<<<gDD, blk, 0, stream>>>(xn_b, wv_b, D, D, bv, nullptr, nullptr, v_b, 0);

    // flash attention
    dim3 gA(S / 128, 2 * 16);
    attn_kernel<<<gA, blk, 0, stream>>>(q_b, k_b, v_b, at_b, S, D);

    // O projection + residual (xn):  y1 = xn + at @ wo^T + bo   (fp32)
    gemm_bf16_kernel<<<gDD, blk, 0, stream>>>(at_b, wo_b, D, D, bo, xn_f, y1_f, nullptr, 0);

    // LN2: y1 -> y2 (fp32 + bf16)
    layernorm_kernel<<<N, 256, 0, stream>>>(y1_f, g2, be2, y2_f, y2_b);

    // FFN1: h = gelu(y2 @ w1^T + b1)  (bf16 out)
    dim3 gDH(N / 128, H / 128);
    gemm_bf16_kernel<<<gDH, blk, 0, stream>>>(y2_b, w1_b, D, H, b1, nullptr, nullptr, h_b, 1);

    // FFN2: out = y2 + h @ w2^T + b2  (fp32 to d_out)
    gemm_bf16_kernel<<<gDD, blk, 0, stream>>>(h_b, w2_b, H, D, b2, y2_f, (float*)d_out, nullptr, 0);
}
